// RBFNet_82970178224745
// MI455X (gfx1250) — matrix operand order repaired and checked by reading, not compiled, not run on hardware
//
#include <hip/hip_runtime.h>
#include <hip/hip_bf16.h>
#include <stdint.h>

// CDNA5 WMMA types
typedef __attribute__((ext_vector_type(16))) __bf16 v16bf;
typedef __attribute__((ext_vector_type(8)))  float  v8f;

union ABreg { v16bf v; uint32_t u[8]; uint4 q[2]; };

static __device__ __forceinline__ uint32_t f2bf(float a) {
    union { float f; uint32_t u; } x; x.f = a;
    return (x.u + 0x7FFFu + ((x.u >> 16) & 1u)) >> 16;   // RNE-ish
}
static __device__ __forceinline__ uint32_t pack_bf(float lo, float hi) {
    return f2bf(lo) | (f2bf(hi) << 16);
}

#define BDIM 8192
#define DDIM 512
#define NDIM 1024
#define CDIM 1000
#define MROWS 32                      // rows per workgroup (2 row-tiles)
#define NTILES (NDIM / 16)            // 64
#define CTILES_PAD 64                 // 1000 cols padded to 64 tiles (zeros)

// ---------------------------------------------------------------------------
// B-fragment layout (bf16, v_wmma_f32_16x16x32_bf16, wave32):
//   dword = tile*(KB*256) + kblock*256 + lane*8 + v
//   holds pair (K = kblock*32 + (lane>>4)*16 + 2v, K+1) of column n = lane&15.
// A-fragment layout (per 16-row slab):
//   dword = kblock*256 + lane*8 + v
//   holds pair (K = kblock*32 + (lane>>4)*16 + 2v, K+1) of row lane&15.
// ---------------------------------------------------------------------------

__global__ __launch_bounds__(256) void conv_mus_kernel(const float* __restrict__ mus,
                                                       uint32_t* __restrict__ frag) {
    const int ntile = blockIdx.x;            // 0..63
    const int t = threadIdx.x;
    const int lane = t >> 3, v = t & 7;
    const int n = ntile * 16 + (lane & 15);
    const int khalf = (lane >> 4) * 16;
    const float* src = mus + (size_t)n * DDIM;
    uint32_t* dst = frag + (size_t)ntile * (16 * 256) + lane * 8 + v;
    #pragma unroll
    for (int kb = 0; kb < 16; ++kb) {
        const int k = kb * 32 + khalf + 2 * v;
        float2 f = *(const float2*)(src + k);
        dst[kb * 256] = pack_bf(f.x, f.y);
    }
}

__global__ __launch_bounds__(256) void conv_w_kernel(const float* __restrict__ W,
                                                     uint32_t* __restrict__ frag) {
    const int ctile = blockIdx.x;            // 0..63 (tile 62 partial, 63 all-zero)
    const int t = threadIdx.x;
    const int lane = t >> 3, v = t & 7;
    const int c = ctile * 16 + (lane & 15);
    const bool valid = (c < CDIM);
    const float* src = W + (size_t)(valid ? c : 0) * NDIM;
    const int khalf = (lane >> 4) * 16;
    uint32_t* dst = frag + (size_t)ctile * (32 * 256) + lane * 8 + v;
    #pragma unroll
    for (int kb = 0; kb < 32; ++kb) {
        uint32_t p = 0;
        if (valid) {
            const int k = kb * 32 + khalf + 2 * v;
            float2 f = *(const float2*)(src + k);
            p = pack_bf(f.x, f.y);
        }
        dst[kb * 256] = p;
    }
}

__global__ __launch_bounds__(256) void musq_kernel(const float* __restrict__ mus,
                                                   const float* __restrict__ sigs,
                                                   float* __restrict__ musq,
                                                   float* __restrict__ scale) {
    const int n = blockIdx.x * 256 + threadIdx.x;    // 4 blocks -> 0..1023
    const float* r = mus + (size_t)n * DDIM;
    float s = 0.0f;
    #pragma unroll 8
    for (int k = 0; k < DDIM; k += 4) {
        float4 f = *(const float4*)(r + k);
        s += f.x * f.x + f.y * f.y + f.z * f.z + f.w * f.w;
    }
    musq[n] = s;
    const float sg = sigs[n];
    scale[n] = -0.5f / (sg * sg);
}

// ---------------------------------------------------------------------------
// Fused main kernel: 32-row slab per workgroup, 8 wave32 waves, 2x2 blocking.
// Dynamic LDS: xbf (2 x 4096 dwords = 32KB) | phib (2 x 8192 dwords = 64KB).
// ---------------------------------------------------------------------------
__global__ __launch_bounds__(256) void
rbf_fused_kernel(const float* __restrict__ x,
                 const uint32_t* __restrict__ musFrag,
                 const float* __restrict__ musq,
                 const float* __restrict__ scale,
                 const uint32_t* __restrict__ wFrag,
                 const float* __restrict__ bias,
                 float* __restrict__ out)
{
    extern __shared__ uint32_t smem[];
    uint32_t* xbf  = smem;               // [mt][kb][lane][v] : 2*16*256 dwords
    uint32_t* phib = smem + 2 * 4096;    // [mt][kb][lane][v] : 2*32*256 dwords
    __shared__ float xsq[MROWS];

    const int t    = threadIdx.x;
    const int wave = t >> 5;
    const int ln   = t & 31;
    const int m    = ln & 15;
    const int half = ln >> 4;
    const int rowbase = blockIdx.x * MROWS;

    // -------- Phase 0: stage x as bf16 A-fragments + |x|^2 --------
    if (t < MROWS) xsq[t] = 0.0f;
    __syncthreads();
    #pragma unroll
    for (int sidx = 0; sidx < 2; ++sidx) {
        const int s   = t + sidx * 256;  // strip 0..511
        const int row = s >> 4;          // 0..31
        const int kb  = s & 15;
        const int mt  = row >> 4;
        const int r16 = row & 15;
        const float* xr = x + (size_t)(rowbase + row) * DDIM + kb * 32;
        float acc = 0.0f;
        #pragma unroll
        for (int j = 0; j < 16; ++j) {
            float2 f = *(const float2*)(xr + 2 * j);
            acc += f.x * f.x + f.y * f.y;
            const int koff = 2 * j;
            const int vv = (koff & 15) >> 1;
            const int hA = koff >> 4;
            xbf[mt * 4096 + kb * 256 + (r16 + 16 * hA) * 8 + vv] = pack_bf(f.x, f.y);
        }
        atomicAdd(&xsq[row], acc);
    }
    __syncthreads();

    float xs[2][8];
    #pragma unroll
    for (int mt = 0; mt < 2; ++mt)
        #pragma unroll
        for (int r = 0; r < 8; ++r) xs[mt][r] = xsq[mt * 16 + r + 8 * half];

    // -------- Phase 1: cross GEMM + Gaussian, phi -> LDS A-fragments --------
    uint16_t* ph16 = (uint16_t*)phib;
    const uint32_t* a0Base = xbf + ln * 8;
    for (int p = wave; p < NTILES / 2; p += 8) {     // tile pair (2p, 2p+1)
        const uint32_t* b0 = musFrag + (size_t)(2 * p) * (16 * 256) + ln * 8;
        const uint32_t* b1 = b0 + 16 * 256;
        v8f acc[2][2] = {{{}, {}}, {{}, {}}};
        #pragma unroll 2
        for (int kb = 0; kb < 16; ++kb) {
            ABreg A0, A1, B0, B1;
            A0.q[0] = *(const uint4*)(a0Base + kb * 256);
            A0.q[1] = *(const uint4*)(a0Base + kb * 256 + 4);
            A1.q[0] = *(const uint4*)(a0Base + 4096 + kb * 256);
            A1.q[1] = *(const uint4*)(a0Base + 4096 + kb * 256 + 4);
            B0.q[0] = *(const uint4*)(b0 + kb * 256);
            B0.q[1] = *(const uint4*)(b0 + kb * 256 + 4);
            B1.q[0] = *(const uint4*)(b1 + kb * 256);
            B1.q[1] = *(const uint4*)(b1 + kb * 256 + 4);
            acc[0][0] = __builtin_amdgcn_wmma_f32_16x16x32_bf16(
                false, A0.v, false, B0.v, (short)0, acc[0][0], false, false);
            acc[0][1] = __builtin_amdgcn_wmma_f32_16x16x32_bf16(
                false, A0.v, false, B1.v, (short)0, acc[0][1], false, false);
            acc[1][0] = __builtin_amdgcn_wmma_f32_16x16x32_bf16(
                false, A1.v, false, B0.v, (short)0, acc[1][0], false, false);
            acc[1][1] = __builtin_amdgcn_wmma_f32_16x16x32_bf16(
                false, A1.v, false, B1.v, (short)0, acc[1][1], false, false);
        }
        #pragma unroll
        for (int j = 0; j < 2; ++j) {
            const int ncol = (2 * p + j) * 16 + m;
            const float mq = musq[ncol];
            const float sc = scale[ncol];
            // phi(m_out, ncol) -> A-frag position for phase-2 GEMM (K dim = n)
            const int kblock = ncol >> 5;
            const int pk  = ncol & 30;
            const int vv  = (pk & 15) >> 1;
            const int hA  = (pk >> 4) & 1;
            const int dwb = kblock * 256 + 16 * hA * 8 + vv;   // + m_out*8
            #pragma unroll
            for (int mt = 0; mt < 2; ++mt) {
                #pragma unroll
                for (int r = 0; r < 8; ++r) {
                    const int m_out = r + 8 * half;
                    float d2 = xs[mt][r] + mq - 2.0f * acc[mt][j][r];
                    d2 = fmaxf(d2, 0.0f);
                    const float pv = __expf(sc * d2);
                    ph16[(mt * 8192 + dwb + m_out * 8) * 2 + (ncol & 1)] =
                        (uint16_t)f2bf(pv);
                }
            }
        }
    }
    __syncthreads();

    // -------- Phase 2: out = phi @ W^T + b --------
    const uint32_t* a2Base = phib + ln * 8;
    for (int p = wave; p < CTILES_PAD / 2; p += 8) { // tile pair (2p, 2p+1)
        const uint32_t* b0 = wFrag + (size_t)(2 * p) * (32 * 256) + ln * 8;
        const uint32_t* b1 = b0 + 32 * 256;
        v8f acc[2][2] = {{{}, {}}, {{}, {}}};
        #pragma unroll 2
        for (int kb = 0; kb < 32; ++kb) {
            ABreg A0, A1, B0, B1;
            A0.q[0] = *(const uint4*)(a2Base + kb * 256);
            A0.q[1] = *(const uint4*)(a2Base + kb * 256 + 4);
            A1.q[0] = *(const uint4*)(a2Base + 8192 + kb * 256);
            A1.q[1] = *(const uint4*)(a2Base + 8192 + kb * 256 + 4);
            B0.q[0] = *(const uint4*)(b0 + kb * 256);
            B0.q[1] = *(const uint4*)(b0 + kb * 256 + 4);
            B1.q[0] = *(const uint4*)(b1 + kb * 256);
            B1.q[1] = *(const uint4*)(b1 + kb * 256 + 4);
            acc[0][0] = __builtin_amdgcn_wmma_f32_16x16x32_bf16(
                false, A0.v, false, B0.v, (short)0, acc[0][0], false, false);
            acc[0][1] = __builtin_amdgcn_wmma_f32_16x16x32_bf16(
                false, A0.v, false, B1.v, (short)0, acc[0][1], false, false);
            acc[1][0] = __builtin_amdgcn_wmma_f32_16x16x32_bf16(
                false, A1.v, false, B0.v, (short)0, acc[1][0], false, false);
            acc[1][1] = __builtin_amdgcn_wmma_f32_16x16x32_bf16(
                false, A1.v, false, B1.v, (short)0, acc[1][1], false, false);
        }
        #pragma unroll
        for (int j = 0; j < 2; ++j) {
            const int c = (2 * p + j) * 16 + m;
            const bool valid = (c < CDIM);
            const float bb = bias[valid ? c : (CDIM - 1)];
            if (valid) {
                #pragma unroll
                for (int mt = 0; mt < 2; ++mt) {
                    #pragma unroll
                    for (int r = 0; r < 8; ++r) {
                        out[(size_t)(rowbase + mt * 16 + r + 8 * half) * CDIM + c] =
                            acc[mt][j][r] + bb;
                    }
                }
            }
        }
    }
}

extern "C" void kernel_launch(void* const* d_in, const int* in_sizes, int n_in,
                              void* d_out, int out_size, void* d_ws, size_t ws_size,
                              hipStream_t stream) {
    const float* x    = (const float*)d_in[0];
    const float* mus  = (const float*)d_in[1];
    const float* sigs = (const float*)d_in[2];
    const float* W    = (const float*)d_in[3];
    const float* b    = (const float*)d_in[4];
    float* out = (float*)d_out;
    (void)in_sizes; (void)n_in; (void)out_size; (void)ws_size;

    // Workspace layout (~3.2 MB total)
    uint8_t* ws = (uint8_t*)d_ws;
    uint32_t* musFrag = (uint32_t*)ws;                        // 64*16*256*4 = 1,048,576 B
    uint32_t* wFrag   = (uint32_t*)(ws + 1048576);            // 64*32*256*4 = 2,097,152 B
    float*    musq    = (float*)(ws + 1048576 + 2097152);     // 4 KB
    float*    scale   = musq + NDIM;                          // 4 KB

    const int dynLds = (2 * 4096 + 2 * 8192) * 4;             // 96 KB
    (void)hipFuncSetAttribute((const void*)rbf_fused_kernel,
                              hipFuncAttributeMaxDynamicSharedMemorySize, dynLds);

    conv_mus_kernel<<<dim3(NTILES), dim3(256), 0, stream>>>(mus, musFrag);
    conv_w_kernel  <<<dim3(CTILES_PAD), dim3(256), 0, stream>>>(W, wFrag);
    musq_kernel    <<<dim3(NDIM / 256), dim3(256), 0, stream>>>(mus, sigs, musq, scale);
    rbf_fused_kernel<<<dim3(BDIM / MROWS), dim3(256), dynLds, stream>>>(
        x, musFrag, musq, scale, wFrag, b, out);
}
